// MultiHeadAttention_59725815218771
// MI455X (gfx1250) — compile-verified, matrix operand order checked by reading, not run on hardware
//
#include <hip/hip_runtime.h>
#include <hip/hip_bf16.h>
#include <stdint.h>

#define D_MODELC 1024
#define NHEADC   16
#define D_KC     64
#define BATCHC   4
#define SEQC     2048
#define MROWSC   (BATCHC * SEQC)   // 8192

typedef __attribute__((ext_vector_type(16))) __bf16 v16bf;
typedef __attribute__((ext_vector_type(8)))  float  v8f;
typedef __attribute__((__vector_size__(4 * sizeof(int)))) int vi4;

union FragAB { v16bf v; uint4 q[2]; };
union AccU   { v8f   v; float f[8]; };

__device__ __forceinline__ unsigned short f2bf(float f) {
  union { __bf16 h; unsigned short s; } u;
  u.h = (__bf16)f;
  return u.s;
}
__device__ __forceinline__ unsigned pack2bf(float a, float b) {
  return (unsigned)f2bf(a) | ((unsigned)f2bf(b) << 16);
}

// ---- 16-lane-row butterfly reductions (VALU via v_permlane16, not DS) ----
#if __has_builtin(__builtin_amdgcn_permlane16)
__device__ __forceinline__ float plx(float x, unsigned s0, unsigned s1) {
  int xi = __builtin_bit_cast(int, x);
  int r = __builtin_amdgcn_permlane16(xi, xi, (int)s0, (int)s1, false, false);
  return __builtin_bit_cast(float, r);
}
__device__ __forceinline__ float red_max16(float x) {
  x = fmaxf(x, plx(x, 0x67452301u, 0xEFCDAB89u));  // xor 1
  x = fmaxf(x, plx(x, 0x54761032u, 0xDCFE98BAu));  // xor 2
  x = fmaxf(x, plx(x, 0x32107654u, 0xBA98FEDCu));  // xor 4
  x = fmaxf(x, plx(x, 0xFEDCBA98u, 0x76543210u));  // xor 8
  return x;
}
__device__ __forceinline__ float red_sum16(float x) {
  x += plx(x, 0x67452301u, 0xEFCDAB89u);
  x += plx(x, 0x54761032u, 0xDCFE98BAu);
  x += plx(x, 0x32107654u, 0xBA98FEDCu);
  x += plx(x, 0xFEDCBA98u, 0x76543210u);
  return x;
}
#else
__device__ __forceinline__ float red_max16(float x) {
  for (int m = 1; m < 16; m <<= 1) x = fmaxf(x, __shfl_xor(x, m, 32));
  return x;
}
__device__ __forceinline__ float red_sum16(float x) {
  for (int m = 1; m < 16; m <<= 1) x += __shfl_xor(x, m, 32);
  return x;
}
#endif

// Load a 16x32 bf16 A/B-style WMMA fragment from LDS/global.
__device__ __forceinline__ v16bf load_frag(const unsigned short* p) {
  FragAB f;
  f.q[0] = *reinterpret_cast<const uint4*>(p);
  f.q[1] = *reinterpret_cast<const uint4*>(p + 16);
  return f.v;
}

__device__ __forceinline__ v8f wmma_bf16(v16bf a, v16bf b, v8f c) {
  return __builtin_amdgcn_wmma_f32_16x16x32_bf16(false, a, false, b, (short)0, c,
                                                 false, false);
}

#if __has_builtin(__builtin_amdgcn_global_load_async_to_lds_b128)
#define HAVE_ASYNC_LDS 1
#endif

// ---------------------------------------------------------------------------
// Tiled GEMM: X[8192, 1024] @ W[1024, 1024] + bias, then *oscale.
// IN_BF16: X is bf16 else f32.
// OUT_MODE 0: f32 [M,1024]
// OUT_MODE 1: bf16 head-split [B, H, L, 64]
// OUT_MODE 2: bf16 head-split transposed [B, H, 64, L]  (for V)
// Block 128x128, BK=64, 8 waves (4x2), wave tile 32x64, 16 WMMA / stage.
// ---------------------------------------------------------------------------
template <bool IN_BF16, int OUT_MODE>
__global__ __launch_bounds__(256) void mha_gemm_kernel(
    const void* __restrict__ Xv, const float* __restrict__ W,
    const float* __restrict__ bias, void* __restrict__ outv, float oscale) {
  __shared__ unsigned short Alds[128 * 72];  // [row][k 0..63], pad to 72
  __shared__ unsigned short Blds[128 * 72];  // transposed [n][k 0..63], pad 72

  const int t    = threadIdx.x;
  const int lane = t & 31;
  const int wave = t >> 5;
  const int hl   = lane >> 4;
  const int lr   = lane & 15;

  const int rowBase = blockIdx.y * 128;
  const int nBase   = blockIdx.x * 128;
  const int wRow    = (wave >> 1) * 32;
  const int wCol    = (wave & 1) * 64;

  v8f acc[2][4];
#pragma unroll
  for (int mt = 0; mt < 2; ++mt)
#pragma unroll
    for (int nt = 0; nt < 4; ++nt) acc[mt][nt] = (v8f){0, 0, 0, 0, 0, 0, 0, 0};

#pragma unroll 1
  for (int kt = 0; kt < D_MODELC / 64; ++kt) {
    const int kBase = kt * 64;
    // ---- stage A (128x64) into LDS as bf16 ----
    if (IN_BF16) {
      const unsigned short* X = (const unsigned short*)Xv;
#pragma unroll
      for (int j = 0; j < 4; ++j) {
        int i = t + 256 * j;
        int r = i >> 3, c8 = (i & 7) * 8;
        const unsigned short* src =
            X + (size_t)(rowBase + r) * D_MODELC + kBase + c8;
        unsigned short* dst = &Alds[r * 72 + c8];
#ifdef HAVE_ASYNC_LDS
        __builtin_amdgcn_global_load_async_to_lds_b128((vi4*)src, (vi4*)dst, 0,
                                                       0);
#else
        *reinterpret_cast<uint4*>(dst) = *reinterpret_cast<const uint4*>(src);
#endif
      }
    } else {
      const float* X = (const float*)Xv;
#pragma unroll
      for (int j = 0; j < 8; ++j) {
        int i = t + 256 * j;
        int r = i >> 4, c4 = (i & 15) * 4;
        float4 v = *reinterpret_cast<const float4*>(
            X + (size_t)(rowBase + r) * D_MODELC + kBase + c4);
        uint2 p; p.x = pack2bf(v.x, v.y); p.y = pack2bf(v.z, v.w);
        *reinterpret_cast<uint2*>(&Alds[r * 72 + c4]) = p;  // b64 store
      }
    }
    // ---- stage B (64x128) transposed into LDS, 4 k-rows packed per store ----
#pragma unroll
    for (int j = 0; j < 2; ++j) {
      int i = t + 256 * j;
      int kq = i >> 5, c4 = (i & 31) * 4;  // k-quad 0..15, n offset
      const float* wp = W + (size_t)(kBase + 4 * kq) * D_MODELC + nBase + c4;
      float4 f0 = *reinterpret_cast<const float4*>(wp);
      float4 f1 = *reinterpret_cast<const float4*>(wp + D_MODELC);
      float4 f2 = *reinterpret_cast<const float4*>(wp + 2 * D_MODELC);
      float4 f3 = *reinterpret_cast<const float4*>(wp + 3 * D_MODELC);
      const float* a0 = (const float*)&f0; const float* a1 = (const float*)&f1;
      const float* a2 = (const float*)&f2; const float* a3 = (const float*)&f3;
#pragma unroll
      for (int cc = 0; cc < 4; ++cc) {
        uint2 p; p.x = pack2bf(a0[cc], a1[cc]); p.y = pack2bf(a2[cc], a3[cc]);
        *reinterpret_cast<uint2*>(&Blds[(c4 + cc) * 72 + 4 * kq]) = p;
      }
    }
    // prefetch next k-tile (global_prefetch_b8)
    if (kt + 1 < D_MODELC / 64) {
      __builtin_prefetch((const char*)W + ((size_t)(kBase + 64 + (t >> 5)) *
                         D_MODELC + nBase + (t & 31) * 4) * 4, 0, 3);
    }
#ifdef HAVE_ASYNC_LDS
    if (IN_BF16) {
#if __has_builtin(__builtin_amdgcn_s_wait_asynccnt)
      __builtin_amdgcn_s_wait_asynccnt(0);
#else
      asm volatile("s_wait_asynccnt 0" ::: "memory");
#endif
    }
#endif
    __syncthreads();

#pragma unroll
    for (int kc = 0; kc < 2; ++kc) {
      v16bf afrag[2];
#pragma unroll
      for (int mt = 0; mt < 2; ++mt)
        afrag[mt] =
            load_frag(&Alds[(wRow + mt * 16 + lr) * 72 + kc * 32 + hl * 8]);
#pragma unroll
      for (int nt = 0; nt < 4; ++nt) {
        v16bf bfrag =
            load_frag(&Blds[(wCol + nt * 16 + lr) * 72 + kc * 32 + hl * 8]);
#pragma unroll
        for (int mt = 0; mt < 2; ++mt)
          acc[mt][nt] = wmma_bf16(afrag[mt], bfrag, acc[mt][nt]);
      }
    }
    __syncthreads();
  }

  // ---- epilogue: bias, scale, store ----
#pragma unroll
  for (int mt = 0; mt < 2; ++mt) {
#pragma unroll
    for (int nt = 0; nt < 4; ++nt) {
      AccU a; a.v = acc[mt][nt];
#pragma unroll
      for (int r = 0; r < 8; ++r) {
        int m = rowBase + wRow + mt * 16 + r + 8 * hl;
        int n = nBase + wCol + nt * 16 + lr;
        float val = (a.f[r] + bias[n]) * oscale;
        if constexpr (OUT_MODE == 0) {
          ((float*)outv)[(size_t)m * D_MODELC + n] = val;
        } else if constexpr (OUT_MODE == 1) {
          int b = m >> 11, q = m & (SEQC - 1);
          int h = n >> 6, d = n & (D_KC - 1);
          ((unsigned short*)outv)[(((size_t)b * NHEADC + h) * SEQC + q) * D_KC +
                                  d] = f2bf(val);
        } else {  // transposed [B,H,64,L] for V
          int b = m >> 11, q = m & (SEQC - 1);
          int h = n >> 6, d = n & (D_KC - 1);
          ((unsigned short*)outv)[(((size_t)b * NHEADC + h) * D_KC + d) * SEQC +
                                  q] = f2bf(val);
        }
      }
    }
  }
}

// ---------------------------------------------------------------------------
// Flash attention, barrier-free: WG = (b*H+h, 128 query rows), 8 waves x 16.
// Q bf16 [B,H,L,64] (pre-scaled by 1/8), K bf16 [B,H,L,64], V bf16 [B,H,64,L].
// K and V fragments stream straight from global (L2-resident per head);
// only per-wave LDS use is the P C-layout -> A-layout transpose.
// ---------------------------------------------------------------------------
__global__ __launch_bounds__(256) void mha_flash_attn_kernel(
    const unsigned short* __restrict__ Q, const unsigned short* __restrict__ K,
    const unsigned short* __restrict__ Vt, unsigned short* __restrict__ Ctx) {
  __shared__ unsigned short Plds[8 * 16 * 72];  // per-wave [m][key], pad 72

  const int t    = threadIdx.x;
  const int lane = t & 31;
  const int wave = t >> 5;
  const int hl   = lane >> 4;
  const int lr   = lane & 15;

  const int bh         = blockIdx.y;
  const int qBase      = blockIdx.x * 128;
  const size_t headOff = (size_t)bh * SEQC * D_KC;

  const unsigned short* qrow =
      Q + headOff + (size_t)(qBase + wave * 16 + lr) * D_KC;
  v16bf qf[2];
#pragma unroll
  for (int c = 0; c < 2; ++c) qf[c] = load_frag(qrow + c * 32 + hl * 8);

  // Per-lane base pointers for the streamed K / V^T fragments.
  const unsigned short* kfr = K + headOff + (size_t)lr * D_KC + hl * 8;
  const unsigned short* vfr = Vt + headOff + (size_t)lr * SEQC + hl * 8;

  v8f o[4];
#pragma unroll
  for (int nc = 0; nc < 4; ++nc) o[nc] = (v8f){0, 0, 0, 0, 0, 0, 0, 0};
  float rmax[8], rsum[8];
#pragma unroll
  for (int r = 0; r < 8; ++r) { rmax[r] = -3.0e38f; rsum[r] = 0.0f; }

  unsigned short* Pw = &Plds[wave * 16 * 72];

#pragma unroll 1
  for (int kb = 0; kb < SEQC / 64; ++kb) {
    const int key0 = kb * 64;

    // ---- S = Q.K^T, 4 key tiles, K B-fragments from global ----
    v8f s[4];
#pragma unroll
    for (int kt2 = 0; kt2 < 4; ++kt2) s[kt2] = (v8f){0, 0, 0, 0, 0, 0, 0, 0};
#pragma unroll
    for (int c = 0; c < 2; ++c) {
#pragma unroll
      for (int kt2 = 0; kt2 < 4; ++kt2) {
        v16bf bfrag =
            load_frag(kfr + (size_t)(key0 + kt2 * 16) * D_KC + c * 32);
        s[kt2] = wmma_bf16(qf[c], bfrag, s[kt2]);
      }
    }

    // ---- online softmax (Q pre-scaled by 1/sqrt(64)) ----
    AccU a0, a1, a2, a3;
    a0.v = s[0]; a1.v = s[1]; a2.v = s[2]; a3.v = s[3];
#pragma unroll
    for (int r = 0; r < 8; ++r) {
      float x0 = a0.f[r], x1 = a1.f[r], x2 = a2.f[r], x3 = a3.f[r];
      float mx = red_max16(fmaxf(fmaxf(x0, x1), fmaxf(x2, x3)));
      float nm    = fmaxf(rmax[r], mx);
      float alpha = __expf(rmax[r] - nm);
      float p0 = __expf(x0 - nm), p1 = __expf(x1 - nm);
      float p2 = __expf(x2 - nm), p3 = __expf(x3 - nm);
      float ps = red_sum16((p0 + p1) + (p2 + p3));
      rsum[r] = rsum[r] * alpha + ps;
      rmax[r] = nm;
#pragma unroll
      for (int nc = 0; nc < 4; ++nc) o[nc][r] *= alpha;
      const int prow = (r + 8 * hl) * 72;
      Pw[prow + lr]      = f2bf(p0);
      Pw[prow + 16 + lr] = f2bf(p1);
      Pw[prow + 32 + lr] = f2bf(p2);
      Pw[prow + 48 + lr] = f2bf(p3);
    }
    // Same-wave LDS RAW (DS is in-order per wave); fence the scheduler.
    asm volatile("s_wait_dscnt 0" ::: "memory");

    // ---- O += P(16x64) . V(64x64), V^T B-fragments from global ----
    v16bf pf[2];
#pragma unroll
    for (int c = 0; c < 2; ++c)
      pf[c] = load_frag(Pw + lr * 72 + c * 32 + hl * 8);
#pragma unroll
    for (int nc = 0; nc < 4; ++nc) {
#pragma unroll
      for (int c = 0; c < 2; ++c) {
        v16bf vf =
            load_frag(vfr + (size_t)(nc * 16) * SEQC + key0 + c * 32);
        o[nc] = wmma_bf16(pf[c], vf, o[nc]);
      }
    }
  }

  // ---- normalize + write context bf16 [B, L, H*64] ----
  const int b = bh >> 4, h = bh & (NHEADC - 1);
#pragma unroll
  for (int nc = 0; nc < 4; ++nc) {
    AccU a; a.v = o[nc];
#pragma unroll
    for (int r = 0; r < 8; ++r) {
      int m = qBase + wave * 16 + r + 8 * hl;
      int d = nc * 16 + lr;
      float val = a.f[r] / rsum[r];
      Ctx[((size_t)b * SEQC + m) * D_MODELC + h * D_KC + d] = f2bf(val);
    }
  }
}

// ---------------------------------------------------------------------------
extern "C" void kernel_launch(void* const* d_in, const int* in_sizes, int n_in,
                              void* d_out, int out_size, void* d_ws,
                              size_t ws_size, hipStream_t stream) {
  const float* query = (const float*)d_in[0];
  const float* key_  = (const float*)d_in[1];
  const float* value = (const float*)d_in[2];
  const float* Wq = (const float*)d_in[3];
  const float* bq = (const float*)d_in[4];
  const float* Wk = (const float*)d_in[5];
  const float* bk = (const float*)d_in[6];
  const float* Wv = (const float*)d_in[7];
  const float* bv = (const float*)d_in[8];
  const float* Wo = (const float*)d_in[9];
  const float* bo = (const float*)d_in[10];
  float* out = (float*)d_out;

  unsigned short* ws = (unsigned short*)d_ws;
  const size_t headElems = (size_t)BATCHC * NHEADC * SEQC * D_KC;  // 8M bf16
  unsigned short* Qws = ws;
  unsigned short* Kws = Qws + headElems;
  unsigned short* Vws = Kws + headElems;   // transposed [B,H,64,L]
  unsigned short* Ctx = Vws + headElems;

  dim3 gblk(D_MODELC / 128, MROWSC / 128);  // (8, 64)
  // Q pre-scaled by 1/sqrt(d_k) so attention needs no score scaling.
  mha_gemm_kernel<false, 1><<<gblk, 256, 0, stream>>>(query, Wq, bq, Qws, 0.125f);
  mha_gemm_kernel<false, 1><<<gblk, 256, 0, stream>>>(key_,  Wk, bk, Kws, 1.0f);
  mha_gemm_kernel<false, 2><<<gblk, 256, 0, stream>>>(value, Wv, bv, Vws, 1.0f);

  dim3 ablk(SEQC / 128, BATCHC * NHEADC);  // (16, 64)
  mha_flash_attn_kernel<<<ablk, 256, 0, stream>>>(Qws, Kws, Vws, Ctx);

  mha_gemm_kernel<true, 0><<<gblk, 256, 0, stream>>>(Ctx, Wo, bo, out, 1.0f);
}
